// DeformConvModule_38328288150092
// MI455X (gfx1250) — matrix operand order repaired and checked by reading, not compiled, not run on hardware
//
#include <hip/hip_runtime.h>
#include <hip/hip_bf16.h>
#include <math.h>

// ---------------------------------------------------------------------------
// CDNA5 (gfx1250) deformable-conv forward, bf16 WMMA path (f32 accumulate).
// ---------------------------------------------------------------------------

typedef __attribute__((ext_vector_type(16))) __bf16 v16bf;
typedef __attribute__((ext_vector_type(8)))  float  v8f;

union FragB16 { uint4 q[2]; unsigned int u[8]; v16bf v; };

#define BB    8
#define CIN   256
#define COUT  256
#define HH    64
#define WW    64

__device__ __forceinline__ unsigned short f2bf(float f) {
  // round-to-nearest-even f32 -> bf16
  unsigned int u = __float_as_uint(f);
  u += 0x7FFFu + ((u >> 16) & 1u);
  return (unsigned short)(u >> 16);
}
__device__ __forceinline__ unsigned int pack2bf(float lo, float hi) {
  return (unsigned int)f2bf(lo) | ((unsigned int)f2bf(hi) << 16);
}
__device__ __forceinline__ float sigmoidf_(float x) {
  return 1.0f / (1.0f + __expf(-x));
}

// ---------------------------------------------------------------------------
// Prep: repack weights into WMMA A-fragment layout (bf16), fold bias+BN.
// A-fragment (16x32 bf16), this code's contiguous K order (B matches):
//   lane<16 : M=lane,    elem j -> K=j      (K 0-15)
//   lane>=16: M=lane-16, elem j -> K=16+j   (K 16-31)
// dcwA[k][cb][mt(16)][lane(32)][j(16)] ; offwA[k][cb][mt(2)][lane][j]
// ---------------------------------------------------------------------------
__global__ __launch_bounds__(256) void prep_kernel(
    const float* __restrict__ dc_w, const float* __restrict__ off_w,
    const float* __restrict__ dc_b,
    const float* __restrict__ bn_g, const float* __restrict__ bn_be,
    const float* __restrict__ bn_m, const float* __restrict__ bn_v,
    unsigned short* __restrict__ dcwA, unsigned short* __restrict__ offwA,
    float* __restrict__ fs, float* __restrict__ fb) {
  const int N_DCW = 9 * 8 * 16 * 32 * 16;  // 589824
  const int N_OFW = 9 * 8 * 2 * 32 * 16;   // 73728
  int i = blockIdx.x * blockDim.x + threadIdx.x;
  if (i < N_DCW) {
    int j = i & 15, lane = (i >> 4) & 31, mt = (i >> 9) & 15, cbk = i >> 13;
    int cb = cbk & 7, k = cbk >> 3;
    int K = ((lane >> 4) << 4) + j;
    int cin = cb * 32 + K;
    int cout = mt * 16 + (lane & 15);
    int kh = k / 3, kw = k % 3;
    dcwA[i] = f2bf(dc_w[((cout * CIN + cin) * 3 + kh) * 3 + kw]);
  } else if (i < N_DCW + N_OFW) {
    int t = i - N_DCW;
    int j = t & 15, lane = (t >> 4) & 31, mt = (t >> 9) & 1, cbk = t >> 10;
    int cb = cbk & 7, k = cbk >> 3;
    int K = ((lane >> 4) << 4) + j;
    int cin = cb * 32 + K;
    int ch = mt * 16 + (lane & 15);
    int kh = k / 3, kw = k % 3;
    float v = (ch < 27) ? off_w[((ch * CIN + cin) * 3 + kh) * 3 + kw] : 0.0f;
    offwA[t] = f2bf(v);
  } else if (i < N_DCW + N_OFW + COUT) {
    int c = i - N_DCW - N_OFW;
    float s = bn_g[c] * rsqrtf(bn_v[c] + 1e-5f);
    fs[c] = s;
    fb[c] = bn_be[c] + (dc_b[c] - bn_m[c]) * s;   // folds conv bias + BN shift
  }
}

// ---------------------------------------------------------------------------
// Pass 1: 3x3 offset conv via WMMA. One block = one (b,h) row: 64 px, 27(->32)
// channels. 8 waves: wave -> (ntile = wave>>1, mtile = wave&1).
// B-fragment layout (32x16 bf16): lane<16 holds K0-15 of col N=lane,
// lane>=16 holds K16-31 of col N=lane-16  -> per-lane 32B contiguous in LDS.
// Staging is branchless: clamped address, v_cndmask zero for padding.
// ---------------------------------------------------------------------------
__global__ __launch_bounds__(256) void offconv_kernel(
    const float* __restrict__ x, const float* __restrict__ off_b,
    const unsigned short* __restrict__ offwA, float* __restrict__ om) {
  __shared__ alignas(16) unsigned short s_b[4][16][32];  // [ntile][N][K]
  const int bid = blockIdx.x;              // b*64 + h
  const int b = bid >> 6, h = bid & 63;
  const int t = threadIdx.x;
  const int lane = t & 31, wave = t >> 5;
  const int nt = wave >> 1, mt = wave & 1;
  // staging assignment: pixel p = t>>2 (0..63), kk0 = (t&3)*8 (8 consecutive K)
  const int sp = t >> 2;
  const int skk0 = (t & 3) << 3;
  unsigned int* sdst = (unsigned int*)&s_b[sp >> 4][sp & 15][skk0];
  const float* xbase = x + ((size_t)(b * CIN + skk0) << 12);
  v8f acc = {};
  const unsigned short* awp =
      offwA + (((size_t)mt * 32 + lane) << 4);  // + (k*8+cb)*1024 per chunk

  for (int k = 0; k < 9; ++k) {
    const int kh = k / 3, kw = k - kh * 3;
    const int yy = h - 1 + kh;
    const bool yok = (yy >= 0) && (yy < HH);
    const int xx = sp - 1 + kw;
    const bool ok = yok && (xx >= 0) && (xx < WW);
    const int yc = min(max(yy, 0), HH - 1);
    const int xc = min(max(xx, 0), WW - 1);
    const int pos = yc * WW + xc;
    for (int cb = 0; cb < 8; ++cb) {
      {  // stage 32K x 64px, branchless, packed b32 LDS stores
        const float* xp = xbase + (((size_t)cb << 5) << 12) + pos;
#pragma unroll
        for (int jj = 0; jj < 4; ++jj) {
          float v0 = xp[(size_t)(2 * jj) << 12];
          float v1 = xp[(size_t)(2 * jj + 1) << 12];
          v0 = ok ? v0 : 0.0f;
          v1 = ok ? v1 : 0.0f;
          sdst[jj] = pack2bf(v0, v1);
        }
      }
      __syncthreads();
      FragB16 bf, af;
      const uint4* bp = (const uint4*)&s_b[nt][lane & 15][(lane >> 4) << 4];
      bf.q[0] = bp[0]; bf.q[1] = bp[1];
      const uint4* ap = (const uint4*)(awp + ((size_t)(k * 8 + cb) << 10));
      af.q[0] = ap[0]; af.q[1] = ap[1];
      acc = __builtin_amdgcn_wmma_f32_16x16x32_bf16(
          false, af.v, false, bf.v, (short)0, acc, false, false);
      __syncthreads();
    }
  }
  const int w = nt * 16 + (lane & 15);
  const int rowoff = (lane >> 4) << 3;
#pragma unroll
  for (int r = 0; r < 8; ++r) {
    int ch = mt * 16 + r + rowoff;
    if (ch < 27) {
      float v = acc[r] + off_b[ch];
      if (ch >= 18) v = sigmoidf_(v);          // mask channels
      om[((size_t)(b * 27 + ch) * HH + h) * WW + w] = v;
    }
  }
}

// ---------------------------------------------------------------------------
// Pass 2: deformable conv. One block = 16 pixels (fixed b,h; w0..w0+15),
// all 256 Cout. 8 waves x 2 M-tiles; shared bilinear-sampled B tile per chunk.
// ---------------------------------------------------------------------------
__global__ __launch_bounds__(256) void deform_kernel(
    const float* __restrict__ x, const float* __restrict__ om,
    const unsigned short* __restrict__ dcwA,
    const float* __restrict__ fs, const float* __restrict__ fb,
    float* __restrict__ out) {
  __shared__ alignas(16) unsigned short s_b[16][32];   // [N][K]
  __shared__ int   s_idx[144][4];                      // [pixel*9+tap]
  __shared__ float s_w[144][4];
  const int bid = blockIdx.x;               // 0..2047
  const int base = bid << 4;
  const int b = base >> 12;
  const int h = (base >> 6) & 63;
  const int w0 = base & 63;
  const int t = threadIdx.x;
  const int lane = t & 31, wave = t >> 5;
  const int mt0 = wave * 2, mt1 = wave * 2 + 1;

  if (t < 144) {  // per (pixel, tap): clamped gather indices + folded weights
    int n = t / 9, k = t - n * 9;
    int kh = k / 3, kw = k - kh * 3;
    int w = w0 + n;
    size_t obase = (((size_t)b * 27) * HH + h) * WW + w;  // channel stride 4096
    float offy = om[obase + (size_t)(2 * k) * 4096];
    float offx = om[obase + (size_t)(2 * k + 1) * 4096];
    float msk  = om[obase + (size_t)(18 + k) * 4096];
    float py = (float)(h - 1 + kh) + offy;
    float px = (float)(w - 1 + kw) + offx;
    float fy = floorf(py), fx = floorf(px);
    float wy1 = py - fy, wx1 = px - fx;
    float wy0 = 1.0f - wy1, wx0 = 1.0f - wx1;
    int y0 = (int)fy, x0 = (int)fx, y1 = y0 + 1, x1 = x0 + 1;
    float vy0 = (y0 >= 0 && y0 < HH) ? 1.0f : 0.0f;
    float vy1 = (y1 >= 0 && y1 < HH) ? 1.0f : 0.0f;
    float vx0 = (x0 >= 0 && x0 < WW) ? 1.0f : 0.0f;
    float vx1 = (x1 >= 0 && x1 < WW) ? 1.0f : 0.0f;
    int yc0 = min(max(y0, 0), HH - 1), yc1 = min(max(y1, 0), HH - 1);
    int xc0 = min(max(x0, 0), WW - 1), xc1 = min(max(x1, 0), WW - 1);
    s_idx[t][0] = yc0 * WW + xc0; s_idx[t][1] = yc0 * WW + xc1;
    s_idx[t][2] = yc1 * WW + xc0; s_idx[t][3] = yc1 * WW + xc1;
    s_w[t][0] = wy0 * wx0 * vy0 * vx0 * msk;
    s_w[t][1] = wy0 * wx1 * vy0 * vx1 * msk;
    s_w[t][2] = wy1 * wx0 * vy1 * vx0 * msk;
    s_w[t][3] = wy1 * wx1 * vy1 * vx1 * msk;
  }
  __syncthreads();

  // staging assignment: pixel p = t>>4 (0..15), kk pair kk0 = 2*(t&15)
  const int sp = t >> 4;
  const int skk0 = (t & 15) << 1;
  unsigned int* sdst = (unsigned int*)&s_b[sp][0];
  const float* xbase = x + ((size_t)(b * CIN + skk0) << 12);

  v8f acc0 = {}, acc1 = {};
  const unsigned short* awp0 = dcwA + (((size_t)mt0 * 32 + lane) << 4);
  const unsigned short* awp1 = dcwA + (((size_t)mt1 * 32 + lane) << 4);

  for (int k = 0; k < 9; ++k) {
    const int ci = sp * 9 + k;
    const int i0 = s_idx[ci][0], i1 = s_idx[ci][1];
    const int i2 = s_idx[ci][2], i3 = s_idx[ci][3];
    const float g0 = s_w[ci][0], g1 = s_w[ci][1];
    const float g2 = s_w[ci][2], g3 = s_w[ci][3];
    for (int cb = 0; cb < 8; ++cb) {
      {  // bilinear-sample 2 adjacent cin for this thread's pixel, pack b32
        const float* xp0 = xbase + (((size_t)cb << 5) << 12);
        const float* xp1 = xp0 + 4096;
        float v0 = g0 * xp0[i0] + g1 * xp0[i1] + g2 * xp0[i2] + g3 * xp0[i3];
        float v1 = g0 * xp1[i0] + g1 * xp1[i1] + g2 * xp1[i2] + g3 * xp1[i3];
        sdst[t & 15] = pack2bf(v0, v1);
      }
      __syncthreads();
      FragB16 bf, a0, a1;
      const uint4* bp = (const uint4*)&s_b[lane & 15][(lane >> 4) << 4];
      bf.q[0] = bp[0]; bf.q[1] = bp[1];
      const size_t choff = (size_t)((k * 8 + cb) * 16) << 9;  // chunk * 8192
      const uint4* ap0 = (const uint4*)(awp0 + choff);
      const uint4* ap1 = (const uint4*)(awp1 + choff);
      a0.q[0] = ap0[0]; a0.q[1] = ap0[1];
      a1.q[0] = ap1[0]; a1.q[1] = ap1[1];
      if (k * 8 + cb + 1 < 72) {  // global_prefetch_b8 next A chunk (linear)
        __builtin_prefetch(awp0 + choff + 8192, 0, 0);
      }
      acc0 = __builtin_amdgcn_wmma_f32_16x16x32_bf16(
          false, a0.v, false, bf.v, (short)0, acc0, false, false);
      acc1 = __builtin_amdgcn_wmma_f32_16x16x32_bf16(
          false, a1.v, false, bf.v, (short)0, acc1, false, false);
      __syncthreads();
    }
  }
  // epilogue: fused BN + ReLU. C layout: VGPR r -> M = r + 8*(lane>=16), N=lane&15
  const int p = lane & 15;
  const int rowoff = (lane >> 4) << 3;
  size_t obase = (((size_t)b * COUT) * HH + h) * WW + (w0 + p);
#pragma unroll
  for (int r = 0; r < 8; ++r) {
    int c0 = mt0 * 16 + r + rowoff;
    out[obase + (size_t)c0 * 4096] = fmaxf(acc0[r] * fs[c0] + fb[c0], 0.0f);
    int c1 = mt1 * 16 + r + rowoff;
    out[obase + (size_t)c1 * 4096] = fmaxf(acc1[r] * fs[c1] + fb[c1], 0.0f);
  }
}

// ---------------------------------------------------------------------------
extern "C" void kernel_launch(void* const* d_in, const int* in_sizes, int n_in,
                              void* d_out, int out_size, void* d_ws, size_t ws_size,
                              hipStream_t stream) {
  const float* x     = (const float*)d_in[0];
  const float* off_w = (const float*)d_in[1];
  const float* off_b = (const float*)d_in[2];
  const float* dc_w  = (const float*)d_in[3];
  const float* dc_b  = (const float*)d_in[4];
  const float* bn_g  = (const float*)d_in[5];
  const float* bn_be = (const float*)d_in[6];
  const float* bn_m  = (const float*)d_in[7];
  const float* bn_v  = (const float*)d_in[8];
  float* out = (float*)d_out;

  char* ws = (char*)d_ws;
  float*          om   = (float*)(ws);                                   // 3,538,944 B
  unsigned short* dcwA = (unsigned short*)(ws + 3538944);                 // 1,179,648 B
  unsigned short* ofwA = (unsigned short*)(ws + 3538944 + 1179648);       //   147,456 B
  float*          fs   = (float*)(ws + 3538944 + 1179648 + 147456);       //     1,024 B
  float*          fb   = fs + 256;                                        //     1,024 B

  prep_kernel<<<2593, 256, 0, stream>>>(dc_w, off_w, dc_b, bn_g, bn_be, bn_m,
                                        bn_v, dcwA, ofwA, fs, fb);
  offconv_kernel<<<512, 256, 0, stream>>>(x, off_b, ofwA, om);
  deform_kernel<<<2048, 256, 0, stream>>>(x, om, dcwA, fs, fb, out);
}
